// SharedFFNBank_28982439313758
// MI455X (gfx1250) — matrix operand order repaired and checked by reading, not compiled, not run on hardware
//
#include <hip/hip_runtime.h>
#include <hip/hip_bf16.h>

typedef __attribute__((ext_vector_type(16))) __bf16 v16bf;
typedef __attribute__((ext_vector_type(8)))  __bf16 v8bf;
typedef __attribute__((ext_vector_type(8)))  float  v8f;

#define T_TOK   1024
#define DMODEL  768
#define HDIM    3072
#define NEXP    8
#define NFLAG   16
#define INDIM   (2*DMODEL + NFLAG)   // 1552
#define HPANEL  512
#define NPANEL  (HDIM / HPANEL)      // 6
#define DPANEL  128
#define NDPAN   (DMODEL / DPANEL)    // 6
#define MAXTILES ((T_TOK + NEXP*16) / 16)  // 72
#define NSLOT   (MAXTILES * 16)            // 1152 padded token slots

// ---------------- EMA: y_t = a*y_{t-1} + (1-a)*x_t ----------------
__global__ void ema_kernel(const float* __restrict__ x, const float* __restrict__ alpha,
                           float* __restrict__ ema) {
  int c = blockIdx.x * blockDim.x + threadIdx.x;
  if (c >= DMODEL) return;
  float a = fminf(fmaxf(alpha[0], 1e-4f), 0.9999f);
  float om = 1.0f - a;
  float y = 0.0f;
  for (int t = 0; t < T_TOK; ++t) {
    y = a * y + om * x[(size_t)t * DMODEL + c];
    ema[(size_t)t * DMODEL + c] = y;
  }
}

// ---------------- init scratch counters ----------------
__global__ void init_kernel(int* __restrict__ counts) {
  if (threadIdx.x < NEXP) counts[threadIdx.x] = 0;
}

// ---------------- router: logits -> tanh cap -> softmax -> top1 ----------------
__global__ void router_kernel(const float* __restrict__ x, const float* __restrict__ ema,
                              const float* __restrict__ flags,
                              const float* __restrict__ rw, const float* __restrict__ rb,
                              float* __restrict__ gate, int* __restrict__ eidx,
                              float* __restrict__ probsAll, int* __restrict__ counts) {
  const int t = blockIdx.x;
  const int wave = threadIdx.x >> 5;   // expert id, 8 waves
  const int lane = threadIdx.x & 31;
  float acc = 0.0f;
  for (int k = lane; k < INDIM; k += 32) {
    float f;
    if (k < DMODEL)            f = x[(size_t)t * DMODEL + k];
    else if (k < 2 * DMODEL)   f = ema[(size_t)t * DMODEL + (k - DMODEL)];
    else                       f = flags[(size_t)t * NFLAG + (k - 2 * DMODEL)];
    acc += f * rw[(size_t)wave * INDIM + k];
  }
  #pragma unroll
  for (int off = 16; off > 0; off >>= 1) acc += __shfl_down(acc, off, 32);
  __shared__ float lg[NEXP];
  if (lane == 0) lg[wave] = acc + rb[wave];
  __syncthreads();
  if (threadIdx.x == 0) {
    float p[NEXP], mx = -1e30f;
    #pragma unroll
    for (int e = 0; e < NEXP; ++e) {
      float l = 30.0f * tanhf(lg[e] * (1.0f / 30.0f));  // TEMP=1, LOGIT_CAP=30
      p[e] = l; mx = fmaxf(mx, l);
    }
    float s = 0.0f;
    #pragma unroll
    for (int e = 0; e < NEXP; ++e) { p[e] = __expf(p[e] - mx); s += p[e]; }
    float inv = 1.0f / s;
    float best = -1.0f; int bi = 0;
    #pragma unroll
    for (int e = 0; e < NEXP; ++e) {
      p[e] *= inv;
      probsAll[(size_t)t * NEXP + e] = p[e];
      if (p[e] > best) { best = p[e]; bi = e; }   // first max (matches argmax)
    }
    gate[t] = best;
    eidx[t] = bi;
    atomicAdd(&counts[bi], 1);
  }
}

// ---------------- build tile-padded expert partition ----------------
__global__ void partition_kernel(const int* __restrict__ counts, int* __restrict__ offsets,
                                 int* __restrict__ tileExpert, int* __restrict__ token_perm,
                                 int* __restrict__ fill) {
  if (threadIdx.x != 0 || blockIdx.x != 0) return;
  int off = 0, ti = 0;
  for (int e = 0; e < NEXP; ++e) {
    offsets[e] = off;
    fill[e] = 0;
    int tiles = (counts[e] + 15) >> 4;
    for (int j = 0; j < tiles; ++j) tileExpert[ti++] = e;
    off += tiles * 16;
  }
  offsets[NEXP] = off;
  for (int i = ti; i < MAXTILES; ++i) tileExpert[i] = -1;
  for (int i = 0; i < NSLOT; ++i) token_perm[i] = -1;
}

__global__ void scatter_kernel(const int* __restrict__ eidx, const int* __restrict__ offsets,
                               int* __restrict__ fill, int* __restrict__ token_perm) {
  int t = blockIdx.x * blockDim.x + threadIdx.x;
  if (t < T_TOK) {
    int e = eidx[t];
    int pos = atomicAdd(&fill[e], 1);
    token_perm[offsets[e] + pos] = t;
  }
}

// ---------------- WMMA fragment helpers (contiguous K order, shared by A and B) ----------------
// A 16x32 bf16: lane (hi = lane>>4, r = lane&15) holds row r;
//   elems 0..15 -> K = kbase + hi*16 + e
__device__ __forceinline__ v16bf frag_a_bf16(const __bf16* __restrict__ rowBase,
                                             int kbase, int hi) {
  v8bf lo = *(const v8bf*)(rowBase + kbase + hi * 16);
  v8bf hb = *(const v8bf*)(rowBase + kbase + hi * 16 + 8);
  v16bf a;
  #pragma unroll
  for (int e = 0; e < 8; ++e) { a[e] = lo[e]; a[8 + e] = hb[e]; }
  return a;
}

// A 16x32 from fp32 row (convert in registers); same K mapping, 16B-aligned float4s
__device__ __forceinline__ v16bf frag_a_f32(const float* __restrict__ rowBase,
                                            int kbase, int hi) {
  const float4* p0 = (const float4*)(rowBase + kbase + hi * 16);
  const float4* p1 = (const float4*)(rowBase + kbase + hi * 16 + 8);
  float4 q0 = p0[0], q1 = p0[1], q2 = p1[0], q3 = p1[1];
  v16bf a;
  a[0]  = (__bf16)q0.x; a[1]  = (__bf16)q0.y; a[2]  = (__bf16)q0.z; a[3]  = (__bf16)q0.w;
  a[4]  = (__bf16)q1.x; a[5]  = (__bf16)q1.y; a[6]  = (__bf16)q1.z; a[7]  = (__bf16)q1.w;
  a[8]  = (__bf16)q2.x; a[9]  = (__bf16)q2.y; a[10] = (__bf16)q2.z; a[11] = (__bf16)q2.w;
  a[12] = (__bf16)q3.x; a[13] = (__bf16)q3.y; a[14] = (__bf16)q3.z; a[15] = (__bf16)q3.w;
  return a;
}

// B 32x16 bf16 from row-major fp32 weights W[n][k]:
// lane holds column n = n0 + r, elems e -> K = kbase + hi*16 + e (16 contiguous floats, 64B aligned)
__device__ __forceinline__ v16bf frag_b_global(const float* __restrict__ p) {
  const float4* p4 = (const float4*)p;
  float4 q0 = p4[0], q1 = p4[1], q2 = p4[2], q3 = p4[3];
  v16bf b;
  b[0]  = (__bf16)q0.x; b[1]  = (__bf16)q0.y; b[2]  = (__bf16)q0.z; b[3]  = (__bf16)q0.w;
  b[4]  = (__bf16)q1.x; b[5]  = (__bf16)q1.y; b[6]  = (__bf16)q1.z; b[7]  = (__bf16)q1.w;
  b[8]  = (__bf16)q2.x; b[9]  = (__bf16)q2.y; b[10] = (__bf16)q2.z; b[11] = (__bf16)q2.w;
  b[12] = (__bf16)q3.x; b[13] = (__bf16)q3.y; b[14] = (__bf16)q3.z; b[15] = (__bf16)q3.w;
  return b;
}

// ---------------- FFN stage 1: H = relu(X W1^T)^2  (bf16 to workspace) ----------------
// grid = MAXTILES * NPANEL; block covers one 16-token tile x one 512-wide h-panel.
__global__ __launch_bounds__(256, 2)
void ffn1_kernel(const float* __restrict__ x, const float* __restrict__ W1,
                 const int* __restrict__ tileExpert, const int* __restrict__ token_perm,
                 __bf16* __restrict__ Hws) {
  const int tile = blockIdx.x / NPANEL;
  const int hp   = blockIdx.x % NPANEL;
  const int te   = tileExpert[tile];
  if (te < 0) return;                       // uniform across block

  const int tid  = threadIdx.x;
  const int wave = tid >> 5;
  const int lane = tid & 31;
  const int r    = lane & 15;
  const int hi   = lane >> 4;

  // A row = token r of this tile; clamp invalid slots to token 0 (their output
  // rows are padded slots never consumed downstream).
  int tok = token_perm[tile * 16 + r];
  const float* xr = x + (size_t)(tok < 0 ? 0 : tok) * DMODEL;
  const float* W1e = W1 + (size_t)te * HDIM * DMODEL;   // [h][d]

  const v8f vzero = {};
  v8f c[4];
  #pragma unroll
  for (int j = 0; j < 4; ++j) c[j] = vzero;

  for (int kc = 0; kc < DMODEL / 32; ++kc) {            // 24 K-chunks
    v16bf a = frag_a_f32(xr, kc * 32, hi);
    #pragma unroll
    for (int j = 0; j < 4; ++j) {                       // 4 of the panel's 32 N-tiles
      int n = hp * HPANEL + (wave + j * 8) * 16 + r;
      v16bf b = frag_b_global(W1e + (size_t)n * DMODEL + kc * 32 + hi * 16);
      c[j] = __builtin_amdgcn_wmma_f32_16x16x32_bf16(false, a, false, b,
                                                     (short)0, c[j], false, false);
    }
  }

  // relu^2 -> Hws[slot][h] bf16
  #pragma unroll
  for (int j = 0; j < 4; ++j) {
    int col = hp * HPANEL + (wave + j * 8) * 16 + r;
    #pragma unroll
    for (int i = 0; i < 8; ++i) {
      int slot = tile * 16 + i + hi * 8;
      float v = c[j][i];
      v = (v > 0.0f) ? v * v : 0.0f;
      Hws[(size_t)slot * HDIM + col] = (__bf16)v;
    }
  }
}

// ---------------- FFN stage 2: out = gate * (H W2^T) ----------------
// grid = MAXTILES * NDPAN; block covers one 16-token tile x one 128-wide d-panel.
__global__ __launch_bounds__(256, 2)
void ffn2_kernel(const __bf16* __restrict__ Hws, const float* __restrict__ W2,
                 const float* __restrict__ gate, const int* __restrict__ tileExpert,
                 const int* __restrict__ token_perm, float* __restrict__ out) {
  const int tile = blockIdx.x / NDPAN;
  const int dp   = blockIdx.x % NDPAN;
  const int te   = tileExpert[tile];
  if (te < 0) return;

  const int tid  = threadIdx.x;
  const int wave = tid >> 5;
  const int lane = tid & 31;
  const int r    = lane & 15;
  const int hi   = lane >> 4;

  const __bf16* hrow = Hws + (size_t)(tile * 16 + r) * HDIM;      // A row (token r)
  const float*  W2e  = W2 + (size_t)te * DMODEL * HDIM;           // [d][h]
  const int n = (dp * 8 + wave) * 16 + r;                         // B column (out dim)
  const float* wrow = W2e + (size_t)n * HDIM;

  // two independent accumulator chains over even/odd K-chunks for WMMA ILP
  v8f acc0 = {}, acc1 = {};
  for (int kc = 0; kc < HDIM / 32; kc += 2) {                     // 96 chunks total
    v16bf a0 = frag_a_bf16(hrow, kc * 32, hi);
    v16bf b0 = frag_b_global(wrow + kc * 32 + hi * 16);
    acc0 = __builtin_amdgcn_wmma_f32_16x16x32_bf16(false, a0, false, b0,
                                                   (short)0, acc0, false, false);
    v16bf a1 = frag_a_bf16(hrow, (kc + 1) * 32, hi);
    v16bf b1 = frag_b_global(wrow + (kc + 1) * 32 + hi * 16);
    acc1 = __builtin_amdgcn_wmma_f32_16x16x32_bf16(false, a1, false, b1,
                                                   (short)0, acc1, false, false);
  }
  v8f acc = acc0 + acc1;

  const int col = (dp * 8 + wave) * 16 + r;
  #pragma unroll
  for (int i = 0; i < 8; ++i) {
    int row = i + hi * 8;
    int tok = token_perm[tile * 16 + row];
    if (tok >= 0) out[(size_t)tok * DMODEL + col] = gate[tok] * acc[i];
  }
}

// ---------------- deterministic load-balance loss ----------------
__global__ void lb_kernel(const float* __restrict__ probsAll, const int* __restrict__ counts,
                          float* __restrict__ out_lb) {
  if (threadIdx.x != 0 || blockIdx.x != 0) return;
  float pmean[NEXP];
  #pragma unroll
  for (int e = 0; e < NEXP; ++e) pmean[e] = 0.0f;
  for (int t = 0; t < T_TOK; ++t)
    for (int e = 0; e < NEXP; ++e) pmean[e] += probsAll[(size_t)t * NEXP + e];
  float s = 0.0f;
  for (int e = 0; e < NEXP; ++e)
    s += ((float)counts[e] * (1.0f / T_TOK)) * (pmean[e] * (1.0f / T_TOK));
  *out_lb = 1e-3f * (float)NEXP * s;   // LB_COEFF * E * sum(frac * pmean)
}

extern "C" void kernel_launch(void* const* d_in, const int* in_sizes, int n_in,
                              void* d_out, int out_size, void* d_ws, size_t ws_size,
                              hipStream_t stream) {
  const float* x     = (const float*)d_in[0];   // [1,1024,768]
  const float* flags = (const float*)d_in[1];   // [1,1024,16]
  const float* W1    = (const float*)d_in[2];   // [8,3072,768]
  const float* W2    = (const float*)d_in[3];   // [8,768,3072]
  const float* rw    = (const float*)d_in[4];   // [8,1552]
  const float* rb    = (const float*)d_in[5];   // [8]
  const float* alpha = (const float*)d_in[6];   // [1]
  float* out = (float*)d_out;                   // [1024*768] + lb scalar

  // workspace carve-up (Hws first for 16B alignment of bf16 rows)
  __bf16* Hws      = (__bf16*)d_ws;                      // 1152*3072 bf16 (7.08 MB)
  float* ema       = (float*)(Hws + (size_t)NSLOT * HDIM);
  float* gate      = ema + (size_t)T_TOK * DMODEL;       // 1024 f32
  int*   eidx      = (int*)(gate + T_TOK);               // 1024 i32
  float* probsAll  = (float*)(eidx + T_TOK);             // 8192 f32
  int*   counts    = (int*)(probsAll + (size_t)T_TOK * NEXP);  // 8
  int*   offsets   = counts + NEXP;                      // 9
  int*   fill      = offsets + NEXP + 1;                 // 8
  int*   tileExp   = fill + NEXP;                        // 72
  int*   tokenPerm = tileExp + MAXTILES;                 // 1152

  init_kernel<<<1, 32, 0, stream>>>(counts);
  ema_kernel<<<(DMODEL + 255) / 256, 256, 0, stream>>>(x, alpha, ema);
  router_kernel<<<T_TOK, 256, 0, stream>>>(x, ema, flags, rw, rb,
                                           gate, eidx, probsAll, counts);
  partition_kernel<<<1, 1, 0, stream>>>(counts, offsets, tileExp, tokenPerm, fill);
  scatter_kernel<<<(T_TOK + 255) / 256, 256, 0, stream>>>(eidx, offsets, fill, tokenPerm);
  ffn1_kernel<<<MAXTILES * NPANEL, 256, 0, stream>>>(x, W1, tileExp, tokenPerm, Hws);
  ffn2_kernel<<<MAXTILES * NDPAN, 256, 0, stream>>>(Hws, W2, gate, tileExp, tokenPerm, out);
  lb_kernel<<<1, 1, 0, stream>>>(probsAll, counts, out + (size_t)T_TOK * DMODEL);
}
